// ModularTreeLSTMPredictor_42649025249412
// MI455X (gfx1250) — matrix operand order repaired and checked by reading, not compiled, not run on hardware
//
#include <hip/hip_runtime.h>

// Problem constants (from reference)
#define B_       128
#define N_       512
#define F_       63
#define H_       128
#define M_       2
#define GATE_IN_ 320           // F+1 + 2*H
#define COLS_    1280          // M * 5H  (both modules computed, selected later)
#define NTILES_  80            // COLS_/16
#define KTILES_  10            // GATE_IN_/32
#define ROWS_    16            // batch tile = WMMA M
#define THREADS_ 512           // 16 waves
#define WAVES_   16
#define TPW_     5             // N-tiles per wave (80/16)

typedef __attribute__((ext_vector_type(16))) _Float16 v16h;
typedef __attribute__((ext_vector_type(8)))  float    v8f;
typedef __attribute__((address_space(1))) const v16h* gv16p;

union AFrag { v16h h; unsigned int u[8]; };

__device__ __forceinline__ float sigf(float x) { return 1.0f / (1.0f + __expf(-x)); }

// Load the 5 B fragments of one k-tile through an explicit addrspace(1)
// pointer so the backend emits global_load_b128 (not flat_load_b128, which
// couples LOADcnt with DScnt and shares the LDS port arbitration).
__device__ __forceinline__ void load_b_frags(const _Float16* wbase, int wave,
                                             int lane, int kt, v16h* dst) {
  #pragma unroll
  for (int i = 0; i < TPW_; ++i) {
    int nt = wave * TPW_ + i;
    unsigned long long addr =
        (unsigned long long)(wbase + (((nt * KTILES_ + kt) << 5) + lane) * 16);
    dst[i] = *(gv16p)addr;
  }
}

// ---------------------------------------------------------------------------
// Pre-swizzle W_gate (M,320,640) f32 -> f16 B-fragments in this file's K order (same as A).
// B matrix 32x16 per fragment:  N = lane&15 ; lanes 0-15 hold K=0..15,
// lanes 16-31 hold K=16..31 ; VGPR v packs (K=2v, K=2v+1).
// Fragment storage: [nt][kt][lane][16 halves] -> contiguous 32B per lane.
// ---------------------------------------------------------------------------
__global__ void prep_weights_kernel(const float* __restrict__ Wg,
                                    _Float16* __restrict__ wsw) {
  int gid = blockIdx.x * blockDim.x + threadIdx.x;
  if (gid >= NTILES_ * KTILES_ * 512) return;
  int frag = gid >> 9;           // which (nt,kt) fragment
  int idx  = gid & 511;          // element within fragment
  int kt   = frag % KTILES_;
  int nt   = frag / KTILES_;
  int lane = idx >> 4;
  int rem  = idx & 15;
  int v    = rem >> 1;
  int hh   = rem & 1;
  int col  = nt * 16 + (lane & 15);           // global column in [0,1280)
  int hi   = lane >> 4;
  int k    = kt * 32 + 2 * v + 16 * hi + hh;  // global K in [0,320)
  int m    = col / 640;
  int o    = col - m * 640;
  wsw[gid] = (_Float16)Wg[(m * GATE_IN_ + k) * 640 + o];
}

// ---------------------------------------------------------------------------
// Main scan kernel: one workgroup owns 16 batch rows, runs all 512 tree steps.
// ---------------------------------------------------------------------------
__global__ void __launch_bounds__(THREADS_)
treelstm_kernel(const float* __restrict__ features,
                const float* __restrict__ treatment,
                const float* __restrict__ b_gate,
                const float* __restrict__ W_out,
                const float* __restrict__ b_out,
                const int*   __restrict__ module_idx,
                const int*   __restrict__ left_child,
                const int*   __restrict__ right_child,
                const _Float16* __restrict__ wsw,
                float* __restrict__ h_buf,
                float* __restrict__ c_buf,
                float* __restrict__ out) {
  __shared__ _Float16 sA[ROWS_ * GATE_IN_];   // 10.0 KB  (A matrix, f16)
  __shared__ float    sG[ROWS_ * COLS_];      // 80.0 KB  (gate outputs, f32)

  const int tid  = threadIdx.x;
  const int lane = tid & 31;
  const int wave = tid >> 5;
  const int b0   = blockIdx.x * ROWS_;
  const int hi   = lane >> 4;       // half-wave select
  const int nrow = lane & 15;       // A: row M ; B/C: column N
  const int b    = b0 + wave;       // this wave's batch row (phases 1 & 4)

  for (int t = 0; t < N_; ++t) {
    const int lc = left_child[t];
    const int rc = right_child[t];

    // Opaque zero offset: blocks LICM from hoisting (and then spilling) the
    // 50 per-step fragment loads, while keeping addrspace info intact.
    unsigned zoff = 0;
    asm volatile("" : "+s"(zoff));
    const _Float16* wbase = wsw + zoff;

    // ---- Phase 1: build combined = [x_t | left_h | right_h] as f16 in LDS.
    // One wave per batch row; lane covers k = e*32 + lane, branches fold
    // per unrolled e since the k range is compile-time known.
    {
      const float* xf = features + ((size_t)b * N_ + t) * F_;
      const float* lh = h_buf + ((size_t)b * N_ + (lc < 0 ? 0 : lc)) * H_;
      const float* rh = h_buf + ((size_t)b * N_ + (rc < 0 ? 0 : rc)) * H_;
      _Float16* sArow = &sA[wave * GATE_IN_];
      #pragma unroll
      for (int e = 0; e < KTILES_; ++e) {
        int k = e * 32 + lane;
        float val;
        if (k < F_)            val = xf[k];
        else if (k == F_)      val = treatment[b];
        else if (k < 64 + H_)  val = (lc >= 0) ? lh[k - 64]  : 0.0f;
        else                   val = (rc >= 0) ? rh[k - 192] : 0.0f;
        sArow[k] = (_Float16)val;
      }
    }
    __syncthreads();

    // ---- Phase 2: (16x320) x (320x1280) GEMM via v_wmma_f32_16x16x32_f16.
    // Explicit double buffering: load k-tile kt+1's B fragments while the
    // five WMMAs of kt execute.
    v8f acc[TPW_] = {};
    v16h bf[2][TPW_];
    load_b_frags(wbase, wave, lane, 0, bf[0]);
    #pragma unroll
    for (int kt = 0; kt < KTILES_; ++kt) {
      // A fragment, K order matching the B fragments:
      //  lanes 0-15: v0..7 -> K=0..15
      //  lanes 16-31: K=16..31 ; VGPR packs (K, K+1)
      AFrag a;
      #pragma unroll
      for (int v = 0; v < 8; ++v) {
        int kp = 2 * v + 16 * hi;
        a.u[v] = *(const unsigned int*)&sA[nrow * GATE_IN_ + kt * 32 + kp];
      }
      if (kt + 1 < KTILES_)
        load_b_frags(wbase, wave, lane, kt + 1, bf[(kt + 1) & 1]);
      #pragma unroll
      for (int i = 0; i < TPW_; ++i) {
        acc[i] = __builtin_amdgcn_wmma_f32_16x16x32_f16(
            false, a.h, false, bf[kt & 1][i], (short)0, acc[i], false, false);
      }
    }

    // ---- Phase 3: scatter C tiles to LDS (VGPR r -> M = r + 8*hi, N = nrow)
    #pragma unroll
    for (int i = 0; i < TPW_; ++i) {
      int nt = wave * TPW_ + i;
      #pragma unroll
      for (int r = 0; r < 8; ++r) {
        sG[(hi * 8 + r) * COLS_ + nt * 16 + nrow] = acc[i][r];
      }
    }
    __syncthreads();

    // ---- Phase 4: LSTM cell math + output dot. One wave per batch row.
    {
      int mod = module_idx[b * N_ + t];
      const float* g  = &sG[wave * COLS_ + mod * 640];
      const float* bg = b_gate + mod * 640;
      float partial = 0.0f;
      #pragma unroll
      for (int jj = 0; jj < 4; ++jj) {
        int j = lane * 4 + jj;
        float ig = g[j]       + bg[j];
        float fl = g[128 + j] + bg[128 + j];
        float fr = g[256 + j] + bg[256 + j];
        float og = g[384 + j] + bg[384 + j];
        float ch = g[512 + j] + bg[512 + j];
        float lcell = (lc >= 0) ? c_buf[((size_t)b * N_ + lc) * H_ + j] : 0.0f;
        float rcell = (rc >= 0) ? c_buf[((size_t)b * N_ + rc) * H_ + j] : 0.0f;
        float cell = sigf(ig) * tanhf(ch) + sigf(fl) * lcell + sigf(fr) * rcell;
        float hid  = sigf(og) * tanhf(cell);
        h_buf[((size_t)b * N_ + t) * H_ + j] = hid;
        c_buf[((size_t)b * N_ + t) * H_ + j] = cell;
        partial += hid * W_out[mod * H_ + j];
      }
      // wave32 reduction across the 32 lanes of this row
      #pragma unroll
      for (int off = 16; off > 0; off >>= 1)
        partial += __shfl_xor(partial, off, 32);
      if (lane == 0) out[b * N_ + t] = partial + b_out[mod];
    }
    __threadfence_block();   // make h/c global stores visible to the WGP
    __syncthreads();
  }
}

// ---------------------------------------------------------------------------
extern "C" void kernel_launch(void* const* d_in, const int* in_sizes, int n_in,
                              void* d_out, int out_size, void* d_ws,
                              size_t ws_size, hipStream_t stream) {
  (void)in_sizes; (void)n_in; (void)out_size; (void)ws_size;
  const float* features   = (const float*)d_in[0];
  const float* treatment  = (const float*)d_in[1];
  const float* W_gate     = (const float*)d_in[2];
  const float* b_gate     = (const float*)d_in[3];
  const float* W_out      = (const float*)d_in[4];
  const float* b_out      = (const float*)d_in[5];
  const int*   module_idx = (const int*)d_in[6];
  const int*   left_child = (const int*)d_in[7];
  const int*   right_child= (const int*)d_in[8];

  // Workspace layout: [0,0.82MB) f16 weight fragments ; then h_buf, c_buf f32
  _Float16* wsw  = (_Float16*)d_ws;
  float* h_buf   = (float*)((char*)d_ws + (1 << 20));
  float* c_buf   = h_buf + (size_t)B_ * N_ * H_;

  prep_weights_kernel<<<(NTILES_ * KTILES_ * 512 + 255) / 256, 256, 0, stream>>>(
      W_gate, wsw);
  treelstm_kernel<<<B_ / ROWS_, THREADS_, 0, stream>>>(
      features, treatment, b_gate, W_out, b_out, module_idx, left_child,
      right_child, wsw, h_buf, c_buf, (float*)d_out);
}
